// BilinearSeqAttnAction_11175504904500
// MI455X (gfx1250) — matrix operand order repaired and checked by reading, not compiled, not run on hardware
//
#include <hip/hip_runtime.h>
#include <hip/hip_bf16.h>
#include <math.h>

// ---------------------------------------------------------------------------
// Problem constants (from reference): B=32, L=4096, XS=YS=256, A=4, C=64, NL=5
// ---------------------------------------------------------------------------
#define BB 32
#define LL 4096
#define XS 256
#define YS 256
#define AA 4
#define CC 64
#define PW 258                    // padded width  (1-pixel zero halo)
#define PPIX (PW * PW)            // 66564 padded pixels
#define WBYTES (9 * CC * CC * 2)  // per-action repacked weights = 73728 B

typedef __attribute__((ext_vector_type(16))) __bf16 v16bf;
typedef __attribute__((ext_vector_type(8)))  float  v8f;
typedef __attribute__((ext_vector_type(8)))  __bf16 bf16x8;

// A-matrix (16x32 bf16) fragment per ISA layout:
//   lane<16  : row M=lane,    K = {0..7, 16..23}
//   lane>=16 : row M=lane-16, K = {8..15, 24..31}
// caller passes base already offset by hiA = (lane>>4)*8
__device__ __forceinline__ v16bf load_a_frag(const __bf16* __restrict__ p) {
    bf16x8 lo = *(const bf16x8*)(p);
    bf16x8 hi = *(const bf16x8*)(p + 16);
    v16bf f;
#pragma unroll
    for (int i = 0; i < 8; ++i) { f[i] = lo[i]; f[i + 8] = hi[i]; }
    return f;
}

// B-matrix (32x16 bf16) fragment, same K order as A:
//   lanes 0-15 hold K={0..7,16..23} of column N=lane, lanes 16-31 hold K={8..15,24..31}.
// caller passes base already offset by (lane>>4)*8.
__device__ __forceinline__ v16bf load_b_frag(const __bf16* __restrict__ p) {
    bf16x8 lo = *(const bf16x8*)(p);
    bf16x8 hi = *(const bf16x8*)(p + 16);
    v16bf f;
#pragma unroll
    for (int i = 0; i < 8; ++i) { f[i] = lo[i]; f[i + 8] = hi[i]; }
    return f;
}

// CDNA5 async global -> LDS copy (ASYNCcnt-tracked, per lane 16B). Inline asm
// per CDNA5 ISA (portable across both toolchains).
__device__ __forceinline__ void async_ld16(unsigned int ldsOff, const void* g) {
    asm volatile("global_load_async_to_lds_b128 %0, %1, off"
                 :: "v"(ldsOff), "v"(g)
                 : "memory");
}
__device__ __forceinline__ void async_wait0() {
    asm volatile("s_wait_asynccnt 0" ::: "memory");
}

// ---------------------------------------------------------------------------
// Zero the 1-pixel halo of a padded activation buffer [A][258][258][64] bf16.
// 1028 halo pixels per action, 8 bf16 per thread -> 32896 threads.
// ---------------------------------------------------------------------------
__global__ __launch_bounds__(256) void halo_zero_kernel(__bf16* __restrict__ act) {
    int idx = blockIdx.x * 256 + threadIdx.x;
    if (idx >= AA * 1028 * 8) return;
    const int c8 = idx & 7;
    const int r  = idx >> 3;
    const int h  = r % 1028;
    const int a  = r / 1028;
    int row, col;
    if (h < 258)      { row = 0;           col = h; }
    else if (h < 516) { row = 257;         col = h - 258; }
    else if (h < 772) { row = h - 516 + 1; col = 0; }
    else              { row = h - 772 + 1; col = 257; }
    bf16x8 z = {};
    *(bf16x8*)(act + ((size_t)a * PPIX + (size_t)row * PW + col) * CC + c8 * 8) = z;
}

// ---------------------------------------------------------------------------
// Weight repack for WMMA layers: cw [A][64][64][3][3] f32 -> [A][9][cout][cin] bf16
// ---------------------------------------------------------------------------
__global__ __launch_bounds__(256) void repack_kernel(const float* __restrict__ cw,
                                                     __bf16* __restrict__ dst) {
    int idx = blockIdx.x * 256 + threadIdx.x;          // ((a*64+co)*64+ci)*9+t
    if (idx >= AA * CC * CC * 9) return;
    int t  = idx % 9;
    int r  = idx / 9;
    int ci = r % CC;
    int r2 = r / CC;
    int co = r2 % CC;
    int a  = r2 / CC;
    dst[(((size_t)a * 9 + t) * CC + co) * CC + ci] = (__bf16)cw[idx];
}

// ---------------------------------------------------------------------------
// Layer 0: Cin=1 -> Cout=64, VALU (75 MFLOP). Writes interior of padded buffer.
// ---------------------------------------------------------------------------
__global__ __launch_bounds__(256) void conv0_kernel(const float* __restrict__ img,  // [256,256]
                                                    const float* __restrict__ w0,   // [A][64][1][3][3]
                                                    const float* __restrict__ b0,   // [A][64]
                                                    __bf16* __restrict__ out) {     // [A][PPIX][64]
    __shared__ float ws[CC * 9];
    __shared__ float bs[CC];
    const int a = blockIdx.y;
    for (int i = threadIdx.x; i < CC * 9; i += 256) ws[i] = w0[a * CC * 9 + i];
    if (threadIdx.x < CC) bs[threadIdx.x] = b0[a * CC + threadIdx.x];
    __syncthreads();

    const int pix = blockIdx.x * 256 + threadIdx.x;
    const int y = pix >> 8, x = pix & 255;
    float v[9];
#pragma unroll
    for (int t = 0; t < 9; ++t) {
        int sy = y + t / 3 - 1, sx = x + t % 3 - 1;
        v[t] = (sy >= 0 && sy < YS && sx >= 0 && sx < XS) ? img[sy * XS + sx] : 0.f;
    }
    __bf16* po = out + ((size_t)a * PPIX + (size_t)(y + 1) * PW + (x + 1)) * CC;
    for (int co = 0; co < CC; ++co) {
        float s = bs[co];
#pragma unroll
        for (int t = 0; t < 9; ++t) s += v[t] * ws[co * 9 + t];
        po[co] = (__bf16)(s > 0.f ? s : 0.f);
    }
}

// ---------------------------------------------------------------------------
// Layers 1..3: 64->64 conv3x3 as 9 shifted channel-GEMMs via bf16 WMMA.
// - 8 waves/block, each wave: one 16-pixel strip x 64 couts.
// - Per-action weights (72KB) staged into LDS once per block with
//   global_load_async_to_lds_b128 + s_wait_asynccnt; B fragments from DS.
// - Tap loop deliberately NOT unrolled: keeps live VGPRs ~100/lane so
//   4 blocks (320KB LDS / 72KB) x 8 waves = 8 waves/SIMD hide LDS+WMMA latency
//   (the fully-unrolled version spilled past 256 VGPRs -> s_set_vgpr_msb,
//   ~3 waves/SIMD).
// ---------------------------------------------------------------------------
__global__ __launch_bounds__(256) void convmid_kernel(const __bf16* __restrict__ in,   // [A][PPIX][64]
                                                      __bf16* __restrict__ out,        // [A][PPIX][64]
                                                      const __bf16* __restrict__ wrep, // [A][9][64][64]
                                                      const float* __restrict__ bias)  // [A][64]
{
    extern __shared__ char smem[];                           // 73728 B dynamic LDS
    const int lane = threadIdx.x & 31;
    const int wave = threadIdx.x >> 5;
    const int task = blockIdx.x * 8 + wave;      // 16384 tasks = A * 256 rows * 16 strips
    const int a    = task >> 12;                 // all 8 waves of a block share `a`
    const int rem  = task & 4095;
    const int y    = rem >> 4;
    const int px0  = (rem & 15) << 4;

    // ---- stage this action's repacked weights into LDS (async copy) ----
    {
        const char* gsrc = (const char*)(wrep + (size_t)a * 9 * CC * CC);
        unsigned int lbase =
            (unsigned int)(size_t)(__attribute__((address_space(3))) char*)smem;
        for (int off = threadIdx.x * 16; off < WBYTES; off += 256 * 16)
            async_ld16(lbase + off, gsrc + off);
        async_wait0();
    }
    __syncthreads();
    const __bf16* wl = (const __bf16*)smem;

    const int m   = lane & 15;                   // A: pixel row; B/D: column (cout)
    const int hiA = (lane >> 4) << 3;            // A fragment half offset (0 or 8)
    const int hiB = (lane >> 4) << 3;            // B fragment half offset (0 or 8)

    const __bf16* inA = in + (size_t)a * PPIX * CC;

    v8f acc[4] = {v8f{}, v8f{}, v8f{}, v8f{}};

#pragma unroll 1
    for (int t = 0; t < 9; ++t) {
        const int dy = t / 3 - 1, dx = t % 3 - 1;
        // padded coords: (y+dy+1, px0+m+dx+1) always in range
        const __bf16* pin =
            inA + ((size_t)(y + dy + 1) * PW + (px0 + m + dx + 1)) * CC;
        const v16bf a0 = load_a_frag(pin + hiA);        // K chunk 0 (cin 0..31)
        const v16bf a1 = load_a_frag(pin + 32 + hiA);   // K chunk 1 (cin 32..63)
        const __bf16* wt = wl + (size_t)t * CC * CC;
#pragma unroll
        for (int n = 0; n < 4; ++n) {
            const __bf16* wc = wt + (size_t)(n * 16 + m) * CC + hiB;
            v16bf b0 = load_b_frag(wc);
            v16bf b1 = load_b_frag(wc + 32);
            acc[n] = __builtin_amdgcn_wmma_f32_16x16x32_bf16(
                false, a0, false, b0, (short)0, acc[n], false, false);
            acc[n] = __builtin_amdgcn_wmma_f32_16x16x32_bf16(
                false, a1, false, b1, (short)0, acc[n], false, false);
        }
    }

    // bias + ReLU + store interior of padded output (channels-last bf16)
    __bf16* outA = out + (size_t)a * PPIX * CC;
#pragma unroll
    for (int n = 0; n < 4; ++n) {
        const float bv = bias[a * CC + n * 16 + m];
#pragma unroll
        for (int r = 0; r < 8; ++r) {
            const int M = r + ((lane >> 4) << 3);           // pixel within strip
            float v = acc[n][r] + bv;
            v = v > 0.f ? v : 0.f;
            outA[((size_t)(y + 1) * PW + (px0 + M + 1)) * CC + n * 16 + m] = (__bf16)v;
        }
    }
}

// ---------------------------------------------------------------------------
// Layer 4: 64 -> 1, no ReLU. Halo is zero so loads are unconditional.
// ---------------------------------------------------------------------------
__global__ __launch_bounds__(256) void conv4_kernel(const __bf16* __restrict__ in,  // [A][PPIX][64]
                                                    const float* __restrict__ w4,   // [A][1][64][3][3]
                                                    const float* __restrict__ b4,   // [A][1]
                                                    float* __restrict__ Wa) {       // [A][65536]
    __shared__ float ws[CC * 9];
    const int a = blockIdx.y;
    for (int i = threadIdx.x; i < CC * 9; i += 256) ws[i] = w4[a * CC * 9 + i];
    __syncthreads();

    const int pix = blockIdx.x * 256 + threadIdx.x;
    const int y = pix >> 8, x = pix & 255;
    float s = b4[a];
#pragma unroll
    for (int t = 0; t < 9; ++t) {
        const int sy = y + t / 3, sx = x + t % 3;           // already +1 padded
        const __bf16* pi = in + ((size_t)a * PPIX + (size_t)sy * PW + sx) * CC;
#pragma unroll
        for (int ci = 0; ci < CC; ++ci) s += (float)pi[ci] * ws[ci * 9 + t];
    }
    Wa[(size_t)a * (XS * YS) + pix] = s;
}

// ---------------------------------------------------------------------------
// Wy[b,x] = sum_h y[b,h] * W_a[actions[b]][h,x]
// ---------------------------------------------------------------------------
__global__ __launch_bounds__(256) void wy_kernel(const float* __restrict__ y,       // [B][256]
                                                 const int* __restrict__ actions,   // [B]
                                                 const float* __restrict__ Wa,      // [A][65536]
                                                 float* __restrict__ Wy) {          // [B][256]
    __shared__ float ys[YS];
    const int b = blockIdx.x;
    ys[threadIdx.x] = y[b * YS + threadIdx.x];
    __syncthreads();
    const float* W = Wa + (size_t)actions[b] * (XS * YS);
    float s = 0.f;
    for (int h = 0; h < YS; ++h) s += ys[h] * W[h * XS + threadIdx.x];
    Wy[b * XS + threadIdx.x] = s;
}

// ---------------------------------------------------------------------------
// xWy[b,l] = dot(x[b,l,:], Wy[b,:]) then mask -> -inf.  One wave per row:
// lane k reads 8 contiguous floats -> wave reads the whole 1KB row coalesced.
// Streams 134 MB of x (the bandwidth floor of this problem).
// ---------------------------------------------------------------------------
__global__ __launch_bounds__(256) void xwy_kernel(const float* __restrict__ x,          // [B][L][XS]
                                                  const unsigned char* __restrict__ mk, // [B][L] bool
                                                  const float* __restrict__ Wy,         // [B][256]
                                                  float* __restrict__ xWy) {            // [B][L]
    __shared__ float wys[XS];
    const int b = blockIdx.y;
    wys[threadIdx.x] = Wy[b * XS + threadIdx.x];
    __syncthreads();

    const int wave = threadIdx.x >> 5, lane = threadIdx.x & 31;
    const int l = blockIdx.x * 8 + wave;
    const float* row = x + ((size_t)b * LL + l) * XS;
    __builtin_prefetch(row + 8 * XS, 0, 0);           // stream ahead (global_prefetch_b8)

    const float4 p0 = *(const float4*)(row + lane * 8);
    const float4 p1 = *(const float4*)(row + lane * 8 + 4);
    const int k = lane * 8;
    float s = p0.x * wys[k + 0] + p0.y * wys[k + 1] + p0.z * wys[k + 2] + p0.w * wys[k + 3]
            + p1.x * wys[k + 4] + p1.y * wys[k + 5] + p1.z * wys[k + 6] + p1.w * wys[k + 7];
#pragma unroll
    for (int off = 16; off > 0; off >>= 1) s += __shfl_xor(s, off, 32);
    if (lane == 0)
        xWy[(size_t)b * LL + l] = mk[(size_t)b * LL + l] ? -INFINITY : s;
}

// ---------------------------------------------------------------------------
// Softmax over L=4096 per batch row.
// ---------------------------------------------------------------------------
__global__ __launch_bounds__(256) void softmax_kernel(const float* __restrict__ xWy,
                                                      float* __restrict__ out) {
    __shared__ float red[256];
    const int b = blockIdx.x;
    const float* v = xWy + (size_t)b * LL;

    float m = -INFINITY;
    for (int i = threadIdx.x; i < LL; i += 256) m = fmaxf(m, v[i]);
    red[threadIdx.x] = m; __syncthreads();
    for (int s = 128; s > 0; s >>= 1) {
        if (threadIdx.x < s) red[threadIdx.x] = fmaxf(red[threadIdx.x], red[threadIdx.x + s]);
        __syncthreads();
    }
    m = red[0];
    if (m == -INFINITY) m = 0.f;                 // degenerate all-masked guard
    __syncthreads();

    float sum = 0.f;
    for (int i = threadIdx.x; i < LL; i += 256) sum += __expf(v[i] - m);
    red[threadIdx.x] = sum; __syncthreads();
    for (int s = 128; s > 0; s >>= 1) {
        if (threadIdx.x < s) red[threadIdx.x] += red[threadIdx.x + s];
        __syncthreads();
    }
    const float inv = 1.f / red[0];
    for (int i = threadIdx.x; i < LL; i += 256)
        out[(size_t)b * LL + i] = __expf(v[i] - m) * inv;
}

// ---------------------------------------------------------------------------
// Host launcher
// ---------------------------------------------------------------------------
extern "C" void kernel_launch(void* const* d_in, const int* in_sizes, int n_in,
                              void* d_out, int out_size, void* d_ws, size_t ws_size,
                              hipStream_t stream) {
    const float*         x       = (const float*)d_in[0];
    const float*         y       = (const float*)d_in[1];
    const unsigned char* x_mask  = (const unsigned char*)d_in[2];  // numpy bool, 1B
    const int*           actions = (const int*)d_in[3];
    const float*         weight  = (const float*)d_in[4];
    const float* cw0 = (const float*)d_in[5];  const float* cb0 = (const float*)d_in[6];
    const float* cw1 = (const float*)d_in[7];  const float* cb1 = (const float*)d_in[8];
    const float* cw2 = (const float*)d_in[9];  const float* cb2 = (const float*)d_in[10];
    const float* cw3 = (const float*)d_in[11]; const float* cb3 = (const float*)d_in[12];
    const float* cw4 = (const float*)d_in[13]; const float* cb4 = (const float*)d_in[14];
    float* out = (float*)d_out;

    // workspace carve-up (256B aligned)
    char*  ws  = (char*)d_ws;
    size_t off = 0;
    auto carve = [&](size_t bytes) -> void* {
        void* p = ws + off;
        off += (bytes + 255) & ~(size_t)255;
        return p;
    };
    const size_t actBytes  = (size_t)AA * PPIX * CC * sizeof(__bf16);  // ~34.1 MB
    const size_t wrepElems = (size_t)AA * 9 * CC * CC;                 // per layer
    __bf16* actA = (__bf16*)carve(actBytes);
    __bf16* actB = (__bf16*)carve(actBytes);
    __bf16* wrep = (__bf16*)carve(3 * wrepElems * sizeof(__bf16));
    float*  Wa   = (float*)carve((size_t)AA * XS * YS * sizeof(float));
    float*  Wy   = (float*)carve((size_t)BB * XS * sizeof(float));
    float*  xWy  = (float*)carve((size_t)BB * LL * sizeof(float));
    (void)ws_size; (void)in_sizes; (void)n_in; (void)out_size;

    // 0) zero the halo border of both padded activation buffers
    {
        const int n = AA * 1028 * 8;
        const int blocks = (n + 255) / 256;
        halo_zero_kernel<<<blocks, 256, 0, stream>>>(actA);
        halo_zero_kernel<<<blocks, 256, 0, stream>>>(actB);
    }

    // 1) repack middle-layer weights to [a][tap][cout][cin] bf16
    {
        const int n = AA * CC * CC * 9;
        const int blocks = (n + 255) / 256;
        repack_kernel<<<blocks, 256, 0, stream>>>(cw1, wrep + 0 * wrepElems);
        repack_kernel<<<blocks, 256, 0, stream>>>(cw2, wrep + 1 * wrepElems);
        repack_kernel<<<blocks, 256, 0, stream>>>(cw3, wrep + 2 * wrepElems);
    }

    // 2) layer 0 (1 -> 64) on the weight image
    conv0_kernel<<<dim3((XS * YS) / 256, AA), 256, 0, stream>>>(weight, cw0, cb0, actA);

    // 3) layers 1..3 via WMMA (ping-pong), weights staged in LDS via async copy
    {
        const int blocks = (AA * YS * (XS / 16)) / 8;   // 8 waves per block
        convmid_kernel<<<blocks, 256, WBYTES, stream>>>(actA, actB, wrep + 0 * wrepElems, cb1);
        convmid_kernel<<<blocks, 256, WBYTES, stream>>>(actB, actA, wrep + 1 * wrepElems, cb2);
        convmid_kernel<<<blocks, 256, WBYTES, stream>>>(actA, actB, wrep + 2 * wrepElems, cb3);
    }

    // 4) layer 4 (64 -> 1, no ReLU) -> W_a
    conv4_kernel<<<dim3((XS * YS) / 256, AA), 256, 0, stream>>>(actB, cw4, cb4, Wa);

    // 5) Wy = y^T * W_a[actions]
    wy_kernel<<<BB, 256, 0, stream>>>(y, actions, Wa, Wy);

    // 6) xWy = x . Wy, masked
    xwy_kernel<<<dim3(LL / 8, BB), 256, 0, stream>>>(x, x_mask, Wy, xWy);

    // 7) softmax over L
    softmax_kernel<<<BB, 256, 0, stream>>>(xWy, out);
}
